// SelfAttention_54142357733654
// MI455X (gfx1250) — matrix operand order repaired and checked by reading, not compiled, not run on hardware
//
#include <hip/hip_runtime.h>
#include <cstdint>

#define DIMN   2048
#define NHEAD  16
#define HDIM   128
#define SEQ    2048
#define BATCH  2
#define ROWS   (BATCH * SEQ)   // 4096
#define NQKV   (3 * DIMN)      // 6144
#define LDT    40              // LDS row stride (ushorts) for 32-wide K tiles (+8 pad)
#define LQ     136             // LDS row stride (ushorts) for 128-wide tiles (+8 pad)

typedef __bf16 bf16_t;
typedef bf16_t v16bf __attribute__((ext_vector_type(16)));
typedef float  v8f   __attribute__((ext_vector_type(8)));

struct __attribute__((aligned(16))) U4 { unsigned x, y, z, w; };

union Frag { v16bf v; unsigned u[8]; unsigned short h[16]; };

__device__ __forceinline__ unsigned short f2bf(float f) {
    union { float f; unsigned u; } c; c.f = f;
    unsigned u = c.u;
    unsigned r = u + 0x7FFFu + ((u >> 16) & 1u);   // round-to-nearest-even
    return (unsigned short)(r >> 16);
}

// CDNA5 async copy: 16 bytes global -> LDS, tracked by ASYNCcnt (no VGPR data).
__device__ __forceinline__ void async_cp16(void* lds, const void* g) {
    asm volatile("global_load_async_to_lds_b128 %0, %1, off"
                 :: "v"((unsigned)(uintptr_t)lds), "v"(g) : "memory");
}
__device__ __forceinline__ void wait_async0() {
    asm volatile("s_wait_asynccnt 0" ::: "memory");
}

// ---------------------------------------------------------------------------
// Pre-pass A: f32 -> bf16 elementwise (x).
// ---------------------------------------------------------------------------
__global__ __launch_bounds__(256) void cvt_bf16(
    const float* __restrict__ in, unsigned short* __restrict__ out)
{
    size_t base = ((size_t)blockIdx.x * 256 + threadIdx.x) * 16;
    alignas(16) unsigned short tmp[16];
    #pragma unroll
    for (int i = 0; i < 16; ++i) tmp[i] = f2bf(in[base + i]);
    *(U4*)&out[base]     = *(const U4*)&tmp[0];
    *(U4*)&out[base + 8] = *(const U4*)&tmp[8];
}

// ---------------------------------------------------------------------------
// Pre-pass B: f32 [R][C] -> bf16 [C][R] (transpose + convert) via 64x64 LDS tile.
// ---------------------------------------------------------------------------
__global__ __launch_bounds__(256) void cvt_transpose(
    const float* __restrict__ in, unsigned short* __restrict__ out, int R, int C)
{
    __shared__ alignas(16) unsigned short tile[64 * 65];
    const int t  = threadIdx.x;
    const int c0 = blockIdx.x * 64, r0 = blockIdx.y * 64;
    #pragma unroll
    for (int i = 0; i < 16; ++i) {
        int idx = t + i * 256;
        int lr = idx >> 6, lc = idx & 63;
        tile[lc * 65 + lr] = f2bf(in[(size_t)(r0 + lr) * C + c0 + lc]);
    }
    __syncthreads();
    #pragma unroll
    for (int i = 0; i < 16; ++i) {
        int idx = t + i * 256;
        int rr = idx >> 6, cc = idx & 63;
        out[(size_t)(c0 + rr) * R + r0 + cc] = tile[rr * 65 + cc];
    }
}

// ---------------------------------------------------------------------------
// Kernel 1: qkv = xb @ Wqt^T + b_qkv ; scatter into Q,K,V bf16 [B,H,S,d].
// Double-buffered async-to-LDS tiles: DMA of step i+1 overlaps WMMA of step i.
// Q is pre-scaled by log2(e)/sqrt(HDIM).
// ---------------------------------------------------------------------------
__global__ __launch_bounds__(256) void qkv_gemm(
    const unsigned short* __restrict__ xb,
    const unsigned short* __restrict__ Wqt,
    const float* __restrict__ bq,
    unsigned short* __restrict__ Qw, unsigned short* __restrict__ Kw,
    unsigned short* __restrict__ Vw)
{
    __shared__ alignas(16) unsigned short a_lds[2][128 * LDT]; // [m][k]
    __shared__ alignas(16) unsigned short b_lds[2][128 * LDT]; // [n][k]

    const int t   = threadIdx.x;
    const int lid = t & 31;
    const int w   = t >> 5;
    const int wm  = w & 3;
    const int wn  = w >> 2;
    const int m0  = blockIdx.y * 128;
    const int n0  = blockIdx.x * 128;

    v8f acc[2][4];
    const v8f vzero = {0.f,0.f,0.f,0.f,0.f,0.f,0.f,0.f};
    #pragma unroll
    for (int sm = 0; sm < 2; ++sm)
        #pragma unroll
        for (int sn = 0; sn < 4; ++sn) acc[sm][sn] = vzero;

    const int r   = t >> 1, seg = (t & 1) * 16;
    const unsigned short* asrc = xb  + (size_t)(m0 + r) * DIMN + seg;
    const unsigned short* bsrc = Wqt + (size_t)(n0 + r) * DIMN + seg;

    // prologue: fill buffer 0
    async_cp16(&a_lds[0][r * LDT + seg],     asrc);
    async_cp16(&a_lds[0][r * LDT + seg + 8], asrc + 8);
    async_cp16(&b_lds[0][r * LDT + seg],     bsrc);
    async_cp16(&b_lds[0][r * LDT + seg + 8], bsrc + 8);

    int cur = 0;
    for (int kk = 0; kk < DIMN; kk += 32) {
        wait_async0();        // this wave's fill of buffer `cur` complete
        __syncthreads();      // all waves filled `cur`; all done computing `cur^1`
        if (kk + 32 < DIMN) { // prefetch next tile into the other buffer
            async_cp16(&a_lds[cur ^ 1][r * LDT + seg],     asrc + kk + 32);
            async_cp16(&a_lds[cur ^ 1][r * LDT + seg + 8], asrc + kk + 40);
            async_cp16(&b_lds[cur ^ 1][r * LDT + seg],     bsrc + kk + 32);
            async_cp16(&b_lds[cur ^ 1][r * LDT + seg + 8], bsrc + kk + 40);
        }

        const unsigned short* al = a_lds[cur];
        const unsigned short* bl = b_lds[cur];
        Frag af[2];
        {
            int h8 = (lid >> 4) * 8;
            #pragma unroll
            for (int sm = 0; sm < 2; ++sm) {
                int m = wm * 32 + sm * 16 + (lid & 15);
                *(U4*)&af[sm].u[0] = *(const U4*)&al[m * LDT + h8];
                *(U4*)&af[sm].u[4] = *(const U4*)&al[m * LDT + 16 + h8];
            }
        }
        #pragma unroll
        for (int sn = 0; sn < 4; ++sn) {
            Frag bfv;
            int n  = wn * 64 + sn * 16 + (lid & 15);
            int ko = (lid >> 4) * 8;
            *(U4*)&bfv.u[0] = *(const U4*)&bl[n * LDT + ko];
            *(U4*)&bfv.u[4] = *(const U4*)&bl[n * LDT + ko + 16];
            #pragma unroll
            for (int sm = 0; sm < 2; ++sm)
                acc[sm][sn] = __builtin_amdgcn_wmma_f32_16x16x32_bf16(
                    false, af[sm].v, false, bfv.v, (short)0, acc[sm][sn],
                    false, false);
        }
        cur ^= 1;
    }

    // ---- epilogue: q/k/v selection and head index are UNIFORM per block
    //      (tile is 128 wide; 3*HDIM = 384 = 3*128).
    const int hh    = n0 / (3 * HDIM);
    const int which = (n0 >> 7) % 3;
    unsigned short* dstp = (which == 0) ? Qw : (which == 1) ? Kw : Vw;
    // fold 1/sqrt(d) * log2(e) into Q so attention softmax uses bare exp2
    const float qs = (which == 0) ? 0.08838834764831845f * 1.4426950408889634f
                                  : 1.0f;
    float bias[4];
    #pragma unroll
    for (int sn = 0; sn < 4; ++sn)
        bias[sn] = bq[n0 + wn * 64 + sn * 16 + (lid & 15)];

    #pragma unroll
    for (int sm = 0; sm < 2; ++sm)
        #pragma unroll
        for (int sn = 0; sn < 4; ++sn)
            #pragma unroll
            for (int i = 0; i < 8; ++i) {
                int rg = m0 + wm * 32 + sm * 16 + i + ((lid >> 4) ? 8 : 0);
                int dc = wn * 64 + sn * 16 + (lid & 15);
                int b = rg >> 11, s = rg & (SEQ - 1);
                float v = (acc[sm][sn][i] + bias[sn]) * qs;
                dstp[(((size_t)b * NHEAD + hh) * SEQ + s) * HDIM + dc] = f2bf(v);
            }
}

// ---------------------------------------------------------------------------
// Kernel 2: causal flash attention per (q-block 128, head, batch).
// Q/K tiles via async-to-LDS DMA; V transposed via bf16-pair packed stores.
// Single-buffered on purpose: 139 KB LDS keeps 2 blocks/WGP for cross-block
// latency hiding. Scores are in log2 units (Q pre-scaled) -> bare exp2.
// ---------------------------------------------------------------------------
__global__ __launch_bounds__(256) void attn(
    const unsigned short* __restrict__ Qw,
    const unsigned short* __restrict__ Kw,
    const unsigned short* __restrict__ Vw,
    unsigned short* __restrict__ Cw)
{
    extern __shared__ __attribute__((aligned(16))) unsigned short smem[];
    unsigned short* q_s  = smem;                 // 128*LQ  [m][d]
    unsigned short* k_s  = q_s  + 128 * LQ;      // 128*LQ  [kpos][d]
    unsigned short* vt_s = k_s  + 128 * LQ;      // 128*LQ  [d][kpos]
    unsigned short* p_s  = vt_s + 128 * LQ;      // 8 waves * 16*LQ

    const int t = threadIdx.x, lid = t & 31, w = t >> 5;
    const int qb = blockIdx.x, h = blockIdx.y, b = blockIdx.z;
    const size_t head_off = (((size_t)b * NHEAD + h) * SEQ) * HDIM;

    // ---- Q tile: async copy (row-major [m][d] == A layout)
    {
        int r = t >> 1, seg = (t & 1) * 64;
        const unsigned short* src = Qw + head_off + (size_t)(qb * 128 + r) * HDIM + seg;
        unsigned short* dst = &q_s[r * LQ + seg];
        #pragma unroll
        for (int i = 0; i < 8; ++i) async_cp16(&dst[i * 8], &src[i * 8]);
    }

    v8f o_acc[8];
    const v8f vzero = {0.f,0.f,0.f,0.f,0.f,0.f,0.f,0.f};
    #pragma unroll
    for (int d = 0; d < 8; ++d) o_acc[d] = vzero;
    float mrow[8], lrow[8];
    #pragma unroll
    for (int i = 0; i < 8; ++i) { mrow[i] = -__builtin_inff(); lrow[i] = 0.f; }

    for (int kb = 0; kb <= qb; ++kb) {
        __syncthreads();   // previous compute done before overwriting K/V tiles
        // ---- K tile: async copy, natural [kpos][d]
        {
            int r = t >> 1, seg = (t & 1) * 64;
            const unsigned short* src = Kw + head_off + (size_t)(kb * 128 + r) * HDIM + seg;
            unsigned short* dst = &k_s[r * LQ + seg];
            #pragma unroll
            for (int i = 0; i < 8; ++i) async_cp16(&dst[i * 8], &src[i * 8]);
        }
        // ---- V tile transposed: vt_s[d][kpos], bf16 pairs packed into b32 stores
        {
            int rp  = (t >> 2) * 2;        // even kpos
            int seg = (t & 3) * 32;        // d segment
            const unsigned short* s0 = Vw + head_off + (size_t)(kb * 128 + rp) * HDIM + seg;
            const unsigned short* s1 = s0 + HDIM;
            alignas(16) unsigned short a0[32], a1[32];
            #pragma unroll
            for (int i = 0; i < 4; ++i) {
                *(U4*)&a0[i * 8] = *(const U4*)&s0[i * 8];
                *(U4*)&a1[i * 8] = *(const U4*)&s1[i * 8];
            }
            #pragma unroll
            for (int e = 0; e < 32; ++e) {
                unsigned pack = (unsigned)a0[e] | ((unsigned)a1[e] << 16);
                *(unsigned*)&vt_s[(seg + e) * LQ + rp] = pack;
            }
        }
        wait_async0();
        __syncthreads();

        // ---- scores: S = Q(16x128) @ K^T(128x128) for this wave's 16 rows
        v8f s_acc[8];
        #pragma unroll
        for (int j = 0; j < 8; ++j) s_acc[j] = vzero;
        #pragma unroll
        for (int kc = 0; kc < 4; ++kc) {
            Frag af;
            int m  = w * 16 + (lid & 15);
            int h8 = (lid >> 4) * 8;
            *(U4*)&af.u[0] = *(const U4*)&q_s[m * LQ + kc * 32 + h8];
            *(U4*)&af.u[4] = *(const U4*)&q_s[m * LQ + kc * 32 + 16 + h8];
            #pragma unroll
            for (int j = 0; j < 8; ++j) {
                Frag bfv;
                int n  = j * 16 + (lid & 15);
                int ko = kc * 32 + (lid >> 4) * 8;
                *(U4*)&bfv.u[0] = *(const U4*)&k_s[n * LQ + ko];
                *(U4*)&bfv.u[4] = *(const U4*)&k_s[n * LQ + ko + 16];
                s_acc[j] = __builtin_amdgcn_wmma_f32_16x16x32_bf16(
                    false, af.v, false, bfv.v, (short)0, s_acc[j], false, false);
            }
        }

        // ---- causal mask on diagonal block
        if (kb == qb) {
            #pragma unroll
            for (int j = 0; j < 8; ++j)
                #pragma unroll
                for (int i = 0; i < 8; ++i) {
                    int gq = w * 16 + i + ((lid >> 4) ? 8 : 0);
                    int gk = j * 16 + (lid & 15);
                    if (gk > gq) s_acc[j][i] = -1.0e30f;
                }
        }

        // ---- online softmax (reductions across the 16 lanes of each half)
        float corr[8];
        #pragma unroll
        for (int i = 0; i < 8; ++i) {
            float rm = s_acc[0][i];
            #pragma unroll
            for (int j = 1; j < 8; ++j) rm = fmaxf(rm, s_acc[j][i]);
            #pragma unroll
            for (int off = 1; off < 16; off <<= 1)
                rm = fmaxf(rm, __shfl_xor(rm, off, 32));
            float mn = fmaxf(mrow[i], rm);
            corr[i] = exp2f(mrow[i] - mn);
            mrow[i] = mn;
            float ps = 0.f;
            #pragma unroll
            for (int j = 0; j < 8; ++j) {
                float p = exp2f(s_acc[j][i] - mn);
                s_acc[j][i] = p;
                ps += p;
            }
            #pragma unroll
            for (int off = 1; off < 16; off <<= 1)
                ps += __shfl_xor(ps, off, 32);
            lrow[i] = lrow[i] * corr[i] + ps;
        }
        #pragma unroll
        for (int d = 0; d < 8; ++d)
            #pragma unroll
            for (int i = 0; i < 8; ++i) o_acc[d][i] *= corr[i];

        // ---- write P (bf16) to wave-private LDS strip in A layout [m][kpos]
        unsigned short* pw = &p_s[w * 16 * LQ];
        #pragma unroll
        for (int j = 0; j < 8; ++j)
            #pragma unroll
            for (int i = 0; i < 8; ++i)
                pw[(i + ((lid >> 4) ? 8 : 0)) * LQ + j * 16 + (lid & 15)] =
                    f2bf(s_acc[j][i]);
        asm volatile("s_wait_dscnt 0" ::: "memory");   // wave-local LDS RAW

        // ---- O += P(16x128) @ V(128x128)
        #pragma unroll
        for (int kc = 0; kc < 4; ++kc) {
            Frag af;
            int h8 = (lid >> 4) * 8;
            *(U4*)&af.u[0] = *(const U4*)&pw[(lid & 15) * LQ + kc * 32 + h8];
            *(U4*)&af.u[4] = *(const U4*)&pw[(lid & 15) * LQ + kc * 32 + 16 + h8];
            #pragma unroll
            for (int d = 0; d < 8; ++d) {
                Frag bfv;
                int n  = d * 16 + (lid & 15);
                int ko = kc * 32 + (lid >> 4) * 8;
                *(U4*)&bfv.u[0] = *(const U4*)&vt_s[n * LQ + ko];
                *(U4*)&bfv.u[4] = *(const U4*)&vt_s[n * LQ + ko + 16];
                o_acc[d] = __builtin_amdgcn_wmma_f32_16x16x32_bf16(
                    false, af.v, false, bfv.v, (short)0, o_acc[d], false, false);
            }
        }
    }

    // ---- normalize and write ctx bf16 [B,S,DIM] (head-major columns)
    #pragma unroll
    for (int d = 0; d < 8; ++d)
        #pragma unroll
        for (int i = 0; i < 8; ++i) {
            int mloc = i + ((lid >> 4) ? 8 : 0);
            int srow = qb * 128 + w * 16 + mloc;
            int col  = h * HDIM + d * 16 + (lid & 15);
            float v = o_acc[d][i] / lrow[i];
            Cw[((size_t)b * SEQ + srow) * DIMN + col] = f2bf(v);
        }
}

// ---------------------------------------------------------------------------
// Kernel 3: out = ctx @ Wot^T + b_out  (double-buffered async tiles, f32 out).
// ---------------------------------------------------------------------------
__global__ __launch_bounds__(256) void out_gemm(
    const unsigned short* __restrict__ Cw,
    const unsigned short* __restrict__ Wot,
    const float* __restrict__ bo, float* __restrict__ out)
{
    __shared__ alignas(16) unsigned short a_lds[2][128 * LDT];
    __shared__ alignas(16) unsigned short b_lds[2][128 * LDT];

    const int t   = threadIdx.x;
    const int lid = t & 31;
    const int w   = t >> 5;
    const int wm  = w & 3;
    const int wn  = w >> 2;
    const int m0  = blockIdx.y * 128;
    const int n0  = blockIdx.x * 128;

    v8f acc[2][4];
    const v8f vzero = {0.f,0.f,0.f,0.f,0.f,0.f,0.f,0.f};
    #pragma unroll
    for (int sm = 0; sm < 2; ++sm)
        #pragma unroll
        for (int sn = 0; sn < 4; ++sn) acc[sm][sn] = vzero;

    const int r   = t >> 1, seg = (t & 1) * 16;
    const unsigned short* asrc = Cw  + (size_t)(m0 + r) * DIMN + seg;
    const unsigned short* bsrc = Wot + (size_t)(n0 + r) * DIMN + seg;

    async_cp16(&a_lds[0][r * LDT + seg],     asrc);
    async_cp16(&a_lds[0][r * LDT + seg + 8], asrc + 8);
    async_cp16(&b_lds[0][r * LDT + seg],     bsrc);
    async_cp16(&b_lds[0][r * LDT + seg + 8], bsrc + 8);

    int cur = 0;
    for (int kk = 0; kk < DIMN; kk += 32) {
        wait_async0();
        __syncthreads();
        if (kk + 32 < DIMN) {
            async_cp16(&a_lds[cur ^ 1][r * LDT + seg],     asrc + kk + 32);
            async_cp16(&a_lds[cur ^ 1][r * LDT + seg + 8], asrc + kk + 40);
            async_cp16(&b_lds[cur ^ 1][r * LDT + seg],     bsrc + kk + 32);
            async_cp16(&b_lds[cur ^ 1][r * LDT + seg + 8], bsrc + kk + 40);
        }

        const unsigned short* al = a_lds[cur];
        const unsigned short* bl = b_lds[cur];
        Frag af[2];
        {
            int h8 = (lid >> 4) * 8;
            #pragma unroll
            for (int sm = 0; sm < 2; ++sm) {
                int m = wm * 32 + sm * 16 + (lid & 15);
                *(U4*)&af[sm].u[0] = *(const U4*)&al[m * LDT + h8];
                *(U4*)&af[sm].u[4] = *(const U4*)&al[m * LDT + 16 + h8];
            }
        }
        #pragma unroll
        for (int sn = 0; sn < 4; ++sn) {
            Frag bfv;
            int n  = wn * 64 + sn * 16 + (lid & 15);
            int ko = (lid >> 4) * 8;
            *(U4*)&bfv.u[0] = *(const U4*)&bl[n * LDT + ko];
            *(U4*)&bfv.u[4] = *(const U4*)&bl[n * LDT + ko + 16];
            #pragma unroll
            for (int sm = 0; sm < 2; ++sm)
                acc[sm][sn] = __builtin_amdgcn_wmma_f32_16x16x32_bf16(
                    false, af[sm].v, false, bfv.v, (short)0, acc[sm][sn],
                    false, false);
        }
        cur ^= 1;
    }

    float bias[4];
    #pragma unroll
    for (int sn = 0; sn < 4; ++sn)
        bias[sn] = bo[n0 + wn * 64 + sn * 16 + (lid & 15)];

    #pragma unroll
    for (int sm = 0; sm < 2; ++sm)
        #pragma unroll
        for (int sn = 0; sn < 4; ++sn)
            #pragma unroll
            for (int i = 0; i < 8; ++i) {
                int rg = m0 + wm * 32 + sm * 16 + i + ((lid >> 4) ? 8 : 0);
                int c  = n0 + wn * 64 + sn * 16 + (lid & 15);
                out[(size_t)rg * DIMN + c] = acc[sm][sn][i] + bias[sn];
            }
}

// ---------------------------------------------------------------------------
extern "C" void kernel_launch(void* const* d_in, const int* in_sizes, int n_in,
                              void* d_out, int out_size, void* d_ws, size_t ws_size,
                              hipStream_t stream)
{
    const float* x  = (const float*)d_in[0];
    const float* Wq = (const float*)d_in[1];
    const float* bq = (const float*)d_in[2];
    const float* Wo = (const float*)d_in[3];
    const float* bo = (const float*)d_in[4];
    float* out = (float*)d_out;

    // workspace layout (bf16 elements)
    unsigned short* xb  = (unsigned short*)d_ws;                 // [4096][2048]
    unsigned short* Wqt = xb  + (size_t)ROWS * DIMN;             // [6144][2048]
    unsigned short* Wot = Wqt + (size_t)NQKV * DIMN;             // [2048][2048]
    unsigned short* Qw  = Wot + (size_t)DIMN * DIMN;             // [B,H,S,d]
    const size_t qkv_elems = (size_t)BATCH * NHEAD * SEQ * HDIM;
    unsigned short* Kw = Qw + qkv_elems;
    unsigned short* Vw = Kw + qkv_elems;
    unsigned short* Cw = Vw + qkv_elems;                         // [B*S][DIM]

    // pre-passes: convert x, transpose+convert weights (one-time, bandwidth-cheap)
    cvt_bf16<<<(ROWS * DIMN) / (16 * 256), 256, 0, stream>>>(x, xb);
    cvt_transpose<<<dim3(NQKV / 64, DIMN / 64), 256, 0, stream>>>(Wq, Wqt, DIMN, NQKV);
    cvt_transpose<<<dim3(DIMN / 64, DIMN / 64), 256, 0, stream>>>(Wo, Wot, DIMN, DIMN);

    qkv_gemm<<<dim3(NQKV / 128, ROWS / 128), 256, 0, stream>>>(xb, Wqt, bq, Qw, Kw, Vw);

    size_t smem = (size_t)(128 * LQ * 3 + 8 * 16 * LQ) * sizeof(unsigned short);
    attn<<<dim3(SEQ / 128, NHEAD, BATCH), 256, smem, stream>>>(Qw, Kw, Vw, Cw);

    out_gemm<<<dim3(DIMN / 128, ROWS / 128), 256, 0, stream>>>(Cw, Wot, bo, out);
}
